// RelHead_63883343560975
// MI455X (gfx1250) — compile-verified, matrix operand order checked by reading, not run on hardware
//
#include <hip/hip_runtime.h>
#include <stdint.h>

typedef __attribute__((ext_vector_type(16))) __bf16 v16bf;
typedef __attribute__((ext_vector_type(8)))  float  v8f;

#define K1     3072
#define FDIM   512
#define VDIM   128
#define NPTS   256
#define SQRTV  11.313708498984761f
#define EPSN   1e-12f

// d_out layout (floats): f [0,1048576) | A [1048576,3145728) | loss [3145728]
#define A_OUT_OFF  1048576
#define LOSS_OFF   3145728

struct Frag8 { unsigned int u[8]; };

__device__ __forceinline__ unsigned int pack_bf2(float a, float b) {
  unsigned int ua = __float_as_uint(a);
  unsigned int ub = __float_as_uint(b);
  ua = (ua + (((ua >> 16) & 1u) + 0x7FFFu)) >> 16;   // RNE f32->bf16
  ub = (ub + (((ub >> 16) & 1u) + 0x7FFFu)) >> 16;
  return ua | (ub << 16);
}
__device__ __forceinline__ unsigned short f2bf(float a) {
  unsigned int ua = __float_as_uint(a);
  return (unsigned short)((ua + (((ua >> 16) & 1u) + 0x7FFFu)) >> 16);
}

// 16x32 bf16 A/B fragment per ISA layout: VGPR0..3 = K=kbase..+7, VGPR4..7 = K=kbase+16..+23
__device__ __forceinline__ v16bf loadFrag(const unsigned int* p, int p0) {
  Frag8 f;
  uint4 lo = *(const uint4*)(p + p0);
  uint4 hi = *(const uint4*)(p + p0 + 8);
  f.u[0] = lo.x; f.u[1] = lo.y; f.u[2] = lo.z; f.u[3] = lo.w;
  f.u[4] = hi.x; f.u[5] = hi.y; f.u[6] = hi.z; f.u[7] = hi.w;
  return __builtin_bit_cast(v16bf, f);
}

// gfx1250 async global->LDS copy (ASYNCcnt-tracked, no VGPR staging)
__device__ __forceinline__ void async_copy_b128(unsigned int lds_byte_off, const void* gptr) {
  asm volatile("global_load_async_to_lds_b128 %0, %1, off"
               :: "v"(lds_byte_off), "v"(gptr) : "memory");
}
__device__ __forceinline__ void wait_async0() {
  asm volatile("s_wait_asynccnt 0x0" ::: "memory");
}

// ---------------- GEMM1: fm = X(8192x3072) * conv_w^T + conv_b -> bf16, double-buffered LDS
__global__ __launch_bounds__(256) void gemm1_patchconv(
    const float* __restrict__ x, const float* __restrict__ convw,
    const float* __restrict__ convb, unsigned short* __restrict__ c1b) {
  __shared__ unsigned int As[2][128][16];   // 2 x (128 x 32 bf16)
  __shared__ unsigned int Bs[2][64][16];    // 2 x (64 x 32 bf16)
  const int tid = threadIdx.x, lane = tid & 31, wv = tid >> 5;
  const int wm = (wv & 3) * 32, wn = (wv >> 2) * 32;
  const int lrow = lane & 15, khalf = lane >> 4;
  const int m0 = blockIdx.x * 128, n0 = blockIdx.y * 64;
  const int arow = tid & 127, ahalf = tid >> 7;  // 16 floats / thread
  const int brow = tid >> 2, bq = tid & 3;       // 8 floats / thread
  const float* aptr = x + (size_t)(m0 + arow) * K1 + ahalf * 16;
  const float* bptr = convw + (size_t)(n0 + brow) * K1 + bq * 8;
  v8f acc[2][2] = {};
  float4 a0, a1, a2, a3, b0, b1;
  // prologue: tile 0 -> regs -> LDS[0]
  a0 = *(const float4*)(aptr + 0);  a1 = *(const float4*)(aptr + 4);
  a2 = *(const float4*)(aptr + 8);  a3 = *(const float4*)(aptr + 12);
  b0 = *(const float4*)(bptr + 0);  b1 = *(const float4*)(bptr + 4);
  {
    unsigned int* dA = &As[0][arow][ahalf * 8];
    dA[0] = pack_bf2(a0.x, a0.y); dA[1] = pack_bf2(a0.z, a0.w);
    dA[2] = pack_bf2(a1.x, a1.y); dA[3] = pack_bf2(a1.z, a1.w);
    dA[4] = pack_bf2(a2.x, a2.y); dA[5] = pack_bf2(a2.z, a2.w);
    dA[6] = pack_bf2(a3.x, a3.y); dA[7] = pack_bf2(a3.z, a3.w);
    unsigned int* dB = &Bs[0][brow][bq * 4];
    dB[0] = pack_bf2(b0.x, b0.y); dB[1] = pack_bf2(b0.z, b0.w);
    dB[2] = pack_bf2(b1.x, b1.y); dB[3] = pack_bf2(b1.z, b1.w);
  }
  for (int it = 0; it < 96; ++it) {
    const int cur = it & 1;
    __syncthreads();                         // tile `it` visible in buf[cur]
    if (it + 1 < 96) {                       // issue next tile's global loads early
      const float* ap = aptr + (it + 1) * 32;
      const float* bp = bptr + (it + 1) * 32;
      a0 = *(const float4*)(ap + 0);  a1 = *(const float4*)(ap + 4);
      a2 = *(const float4*)(ap + 8);  a3 = *(const float4*)(ap + 12);
      b0 = *(const float4*)(bp + 0);  b1 = *(const float4*)(bp + 4);
    }
    const int p0 = khalf * 4;
    v16bf af0 = loadFrag(&As[cur][wm + lrow][0], p0);
    v16bf af1 = loadFrag(&As[cur][wm + 16 + lrow][0], p0);
    v16bf bf0 = loadFrag(&Bs[cur][wn + lrow][0], p0);
    v16bf bf1 = loadFrag(&Bs[cur][wn + 16 + lrow][0], p0);
    acc[0][0] = __builtin_amdgcn_wmma_f32_16x16x32_bf16(false, af0, false, bf0, (short)0, acc[0][0], false, false);
    acc[0][1] = __builtin_amdgcn_wmma_f32_16x16x32_bf16(false, af0, false, bf1, (short)0, acc[0][1], false, false);
    acc[1][0] = __builtin_amdgcn_wmma_f32_16x16x32_bf16(false, af1, false, bf0, (short)0, acc[1][0], false, false);
    acc[1][1] = __builtin_amdgcn_wmma_f32_16x16x32_bf16(false, af1, false, bf1, (short)0, acc[1][1], false, false);
    if (it + 1 < 96) {                       // convert + store into alternate buffer
      unsigned int* dA = &As[1 - cur][arow][ahalf * 8];
      dA[0] = pack_bf2(a0.x, a0.y); dA[1] = pack_bf2(a0.z, a0.w);
      dA[2] = pack_bf2(a1.x, a1.y); dA[3] = pack_bf2(a1.z, a1.w);
      dA[4] = pack_bf2(a2.x, a2.y); dA[5] = pack_bf2(a2.z, a2.w);
      dA[6] = pack_bf2(a3.x, a3.y); dA[7] = pack_bf2(a3.z, a3.w);
      unsigned int* dB = &Bs[1 - cur][brow][bq * 4];
      dB[0] = pack_bf2(b0.x, b0.y); dB[1] = pack_bf2(b0.z, b0.w);
      dB[2] = pack_bf2(b1.x, b1.y); dB[3] = pack_bf2(b1.z, b1.w);
    }
  }
#pragma unroll
  for (int mt = 0; mt < 2; ++mt)
#pragma unroll
    for (int nt = 0; nt < 2; ++nt) {
      const int N = n0 + wn + nt * 16 + lrow;
      const float bias = convb[N];
#pragma unroll
      for (int r = 0; r < 8; ++r) {
        const int M = m0 + wm + mt * 16 + r + khalf * 8;
        c1b[(size_t)M * FDIM + N] = f2bf(acc[mt][nt][r] + bias);
      }
    }
}

// ---------------- GEMM2: f = fm(bf16) * fc1_w^T + fc1_b -> f32
// A tiles streamed with gfx1250 async global->LDS copies (ASYNCcnt), double-buffered.
__global__ __launch_bounds__(256) void gemm2_fc(
    const unsigned short* __restrict__ c1b, const float* __restrict__ fc1w,
    const float* __restrict__ fc1b, float* __restrict__ fout) {
  __shared__ unsigned int As[2][128][16];
  __shared__ unsigned int Bs[2][64][16];
  const int tid = threadIdx.x, lane = tid & 31, wv = tid >> 5;
  const int wm = (wv & 3) * 32, wn = (wv >> 2) * 32;
  const int lrow = lane & 15, khalf = lane >> 4;
  const int m0 = blockIdx.x * 128, n0 = blockIdx.y * 64;
  const int brow = tid >> 2, bq = tid & 3;
  // A tile = 128 rows x 64 bytes = 512 x b128 chunks; this thread owns chunks 2*tid, 2*tid+1
  const int c0 = tid * 2, c1 = tid * 2 + 1;
  const int r0 = c0 >> 2, q0 = c0 & 3, r1 = c1 >> 2, q1 = c1 & 3;
  const unsigned int ldsA = (unsigned int)(unsigned long long)(uintptr_t)&As[0][0][0];
  const unsigned int off0 = r0 * 64 + q0 * 16, off1 = r1 * 64 + q1 * 16;
  const unsigned short* gA0 = c1b + (size_t)(m0 + r0) * FDIM + q0 * 8;
  const unsigned short* gA1 = c1b + (size_t)(m0 + r1) * FDIM + q1 * 8;
  const float* bptr = fc1w + (size_t)(n0 + brow) * FDIM + bq * 8;
  v8f acc[2][2] = {};
  // prologue: async tile 0 -> As[0], B tile 0 -> regs
  async_copy_b128(ldsA + off0, gA0);
  async_copy_b128(ldsA + off1, gA1);
  float4 b0 = *(const float4*)(bptr + 0), b1 = *(const float4*)(bptr + 4);
  for (int ks = 0; ks < 16; ++ks) {
    const int cur = ks & 1;
    wait_async0();                           // own piece of tile ks landed in As[cur]
    unsigned int* dB = &Bs[cur][brow][bq * 4];
    dB[0] = pack_bf2(b0.x, b0.y); dB[1] = pack_bf2(b0.z, b0.w);
    dB[2] = pack_bf2(b1.x, b1.y); dB[3] = pack_bf2(b1.z, b1.w);
    __syncthreads();                         // tile ks (A async + B stores) visible to all waves
    if (ks + 1 < 16) {                       // prefetch tile ks+1 while computing tile ks
      const unsigned int dst = ldsA + (1 - cur) * 8192;
      async_copy_b128(dst + off0, gA0 + (ks + 1) * 32);
      async_copy_b128(dst + off1, gA1 + (ks + 1) * 32);
      const float* bp = bptr + (ks + 1) * 32;
      b0 = *(const float4*)(bp + 0); b1 = *(const float4*)(bp + 4);
    }
    const int p0 = khalf * 4;
    v16bf af0 = loadFrag(&As[cur][wm + lrow][0], p0);
    v16bf af1 = loadFrag(&As[cur][wm + 16 + lrow][0], p0);
    v16bf bf0 = loadFrag(&Bs[cur][wn + lrow][0], p0);
    v16bf bf1 = loadFrag(&Bs[cur][wn + 16 + lrow][0], p0);
    acc[0][0] = __builtin_amdgcn_wmma_f32_16x16x32_bf16(false, af0, false, bf0, (short)0, acc[0][0], false, false);
    acc[0][1] = __builtin_amdgcn_wmma_f32_16x16x32_bf16(false, af0, false, bf1, (short)0, acc[0][1], false, false);
    acc[1][0] = __builtin_amdgcn_wmma_f32_16x16x32_bf16(false, af1, false, bf0, (short)0, acc[1][0], false, false);
    acc[1][1] = __builtin_amdgcn_wmma_f32_16x16x32_bf16(false, af1, false, bf1, (short)0, acc[1][1], false, false);
    __syncthreads();                         // done reading before next iter overwrites buffers
  }
#pragma unroll
  for (int mt = 0; mt < 2; ++mt)
#pragma unroll
    for (int nt = 0; nt < 2; ++nt) {
      const int N = n0 + wn + nt * 16 + lrow;
      const float bias = fc1b[N];
#pragma unroll
      for (int r = 0; r < 8; ++r) {
        const int M = m0 + wm + mt * 16 + r + khalf * 8;
        fout[(size_t)M * VDIM + N] = acc[mt][nt][r] + bias;
      }
    }
}

// ---------------- Row L2 normalize: f(8192x128,f32) -> fnorm(bf16)
__global__ __launch_bounds__(256) void rownorm(
    const float* __restrict__ f, unsigned int* __restrict__ fnu) {
  const int tid = threadIdx.x, lane = tid & 31, wv = tid >> 5;
  const int row = blockIdx.x * 8 + wv;
  const float* rp = f + (size_t)row * VDIM + lane * 4;
  float4 v = *(const float4*)rp;
  float ss = v.x * v.x + v.y * v.y + v.z * v.z + v.w * v.w;
  for (int off = 16; off > 0; off >>= 1) ss += __shfl_xor(ss, off);
  const float scale = 1.0f / fmaxf(sqrtf(ss), EPSN);
  uint2 o;
  o.x = pack_bf2(v.x * scale, v.y * scale);
  o.y = pack_bf2(v.z * scale, v.w * scale);
  *(uint2*)(fnu + (size_t)row * (VDIM / 2) + lane * 2) = o;
}

// ---------------- GEMM3 (batched): A[b] = Fn_b(256x128) * Fn_b^T -> f32 (L2-resident operand)
__global__ __launch_bounds__(256) void gemm3_affinity(
    const unsigned int* __restrict__ fnu, float* __restrict__ outA) {
  const int tid = threadIdx.x, lane = tid & 31, wv = tid >> 5;
  const int wm = (wv & 3) * 32, wn = (wv >> 2) * 32;
  const int lrow = lane & 15, khalf = lane >> 4;
  const int m0 = blockIdx.x * 128, n0 = blockIdx.y * 64, b = blockIdx.z;
  const unsigned int* base = fnu + (size_t)b * NPTS * (VDIM / 2);
  v8f acc[2][2] = {};
#pragma unroll
  for (int ks = 0; ks < 4; ++ks) {
    const int ku = ks * 16 + khalf * 4;
    v16bf af0 = loadFrag(base + (size_t)(m0 + wm + lrow) * 64, ku);
    v16bf af1 = loadFrag(base + (size_t)(m0 + wm + 16 + lrow) * 64, ku);
    v16bf bf0 = loadFrag(base + (size_t)(n0 + wn + lrow) * 64, ku);
    v16bf bf1 = loadFrag(base + (size_t)(n0 + wn + 16 + lrow) * 64, ku);
    acc[0][0] = __builtin_amdgcn_wmma_f32_16x16x32_bf16(false, af0, false, bf0, (short)0, acc[0][0], false, false);
    acc[0][1] = __builtin_amdgcn_wmma_f32_16x16x32_bf16(false, af0, false, bf1, (short)0, acc[0][1], false, false);
    acc[1][0] = __builtin_amdgcn_wmma_f32_16x16x32_bf16(false, af1, false, bf0, (short)0, acc[1][0], false, false);
    acc[1][1] = __builtin_amdgcn_wmma_f32_16x16x32_bf16(false, af1, false, bf1, (short)0, acc[1][1], false, false);
  }
  float* ob = outA + (size_t)b * NPTS * NPTS;
#pragma unroll
  for (int mt = 0; mt < 2; ++mt)
#pragma unroll
    for (int nt = 0; nt < 2; ++nt) {
      const int N = n0 + wn + nt * 16 + lrow;
#pragma unroll
      for (int r = 0; r < 8; ++r) {
        const int M = m0 + wm + mt * 16 + r + khalf * 8;
        ob[(size_t)M * NPTS + N] = acc[mt][nt][r];
      }
    }
}

// ---------------- Column softmax stats for A1 (softmax over dim -2, per column j)
__global__ __launch_bounds__(256) void colstats(
    const float* __restrict__ A, const float* __restrict__ drop1,
    float* __restrict__ M1, float* __restrict__ Z1, float* __restrict__ invS1) {
  const int tid = threadIdx.x, lane = tid & 31, wv = tid >> 5;
  const int g = blockIdx.x * 8 + wv;
  const int b = g >> 8, j = g & 255;
  const float* Ab = A + (size_t)b * NPTS * NPTS;
  const float* Db = drop1 + (size_t)b * NPTS * NPTS;
  float va[8];
#pragma unroll
  for (int it = 0; it < 8; ++it) {
    const int k = it * 32 + lane;
    va[it] = Ab[(size_t)k * NPTS + j] * SQRTV;
  }
  float m = va[0];
#pragma unroll
  for (int it = 1; it < 8; ++it) m = fmaxf(m, va[it]);
  for (int off = 16; off > 0; off >>= 1) m = fmaxf(m, __shfl_xor(m, off));
  float z = 0.f, num = 0.f;
#pragma unroll
  for (int it = 0; it < 8; ++it) {
    const int k = it * 32 + lane;
    const float e = expf(va[it] - m);
    z += e;
    num += (k == j) ? 0.f : e * Db[(size_t)k * NPTS + j];
  }
  for (int off = 16; off > 0; off >>= 1) { z += __shfl_xor(z, off); num += __shfl_xor(num, off); }
  if (lane == 0) { M1[g] = m; Z1[g] = z; invS1[g] = 1.f / fmaxf(num / z, EPSN); }
}

// ---------------- Row softmax stats for A2t (per row of Atemp)
__global__ __launch_bounds__(256) void rowstats(
    const float* __restrict__ A, const float* __restrict__ drop2,
    float* __restrict__ M2, float* __restrict__ Z2, float* __restrict__ invT) {
  const int tid = threadIdx.x, lane = tid & 31, wv = tid >> 5;
  const int g = blockIdx.x * 8 + wv;
  const int b = g >> 8, row = g & 255;
  const float* Ab = A + (size_t)b * NPTS * NPTS + (size_t)row * NPTS;
  const float* Db = drop2 + (size_t)b * NPTS * NPTS;
  float va[8];
#pragma unroll
  for (int it = 0; it < 8; ++it) {
    const int i = it * 32 + lane;
    va[it] = Ab[i] * SQRTV;
  }
  float m = va[0];
#pragma unroll
  for (int it = 1; it < 8; ++it) m = fmaxf(m, va[it]);
  for (int off = 16; off > 0; off >>= 1) m = fmaxf(m, __shfl_xor(m, off));
  float z = 0.f, num = 0.f;
#pragma unroll
  for (int it = 0; it < 8; ++it) {
    const int i = it * 32 + lane;
    const float e = expf(va[it] - m);
    z += e;
    num += (i == row) ? 0.f : e * Db[(size_t)i * NPTS + row];
  }
  for (int off = 16; off > 0; off >>= 1) { z += __shfl_xor(z, off); num += __shfl_xor(num, off); }
  if (lane == 0) { M2[g] = m; Z2[g] = z; invT[g] = 1.f / fmaxf(num / z, EPSN); }
}

// ---------------- diag[b,i] = sum_k A2t[b,i,k] * A1[b,k,i]
__global__ __launch_bounds__(256) void diagk(
    const float* __restrict__ A, const float* __restrict__ drop1,
    const float* __restrict__ drop2,
    const float* __restrict__ M1, const float* __restrict__ Z1, const float* __restrict__ invS1,
    const float* __restrict__ M2, const float* __restrict__ Z2, const float* __restrict__ invT,
    float* __restrict__ diag) {
  const int tid = threadIdx.x, lane = tid & 31, wv = tid >> 5;
  const int g = blockIdx.x * 8 + wv;
  const int b = g >> 8, i = g & 255;
  const float* Ab = A + (size_t)b * NPTS * NPTS;
  const float* D1 = drop1 + (size_t)b * NPTS * NPTS;
  const float* D2 = drop2 + (size_t)b * NPTS * NPTS + (size_t)i * NPTS;
  const float Mi = M1[g], Zi = Z1[g], iSi = invS1[g];
  float acc = 0.f;
#pragma unroll
  for (int it = 0; it < 8; ++it) {
    const int k = it * 32 + lane;
    const float at = Ab[(size_t)k * NPTS + i] * SQRTV;
    const float e1 = expf(at - Mi) / Zi;
    const int gk = (b << 8) + k;
    const float e2 = expf(at - M2[gk]) / Z2[gk];
    const float t = (k == i) ? 0.f
        : e1 * D1[(size_t)k * NPTS + i] * iSi * e2 * D2[k] * invT[gk];
    acc += t;
  }
  for (int off = 16; off > 0; off >>= 1) acc += __shfl_xor(acc, off);
  if (lane == 0) diag[g] = acc;
}

// ---------------- loss = -mean(diag), single block, deterministic
__global__ __launch_bounds__(256) void lossk(const float* __restrict__ diag, float* __restrict__ out) {
  __shared__ float sm[256];
  const int tid = threadIdx.x;
  float s = 0.f;
#pragma unroll
  for (int v = 0; v < 32; ++v) s += diag[tid + 256 * v];
  sm[tid] = s;
  __syncthreads();
  for (int off = 128; off > 0; off >>= 1) {
    if (tid < off) sm[tid] += sm[tid + off];
    __syncthreads();
  }
  if (tid == 0) out[LOSS_OFF] = -sm[0] / 8192.f;
}

extern "C" void kernel_launch(void* const* d_in, const int* in_sizes, int n_in,
                              void* d_out, int out_size, void* d_ws, size_t ws_size,
                              hipStream_t stream) {
  const float* x     = (const float*)d_in[0];
  const float* convw = (const float*)d_in[1];
  const float* convb = (const float*)d_in[2];
  const float* fc1w  = (const float*)d_in[3];
  const float* fc1b  = (const float*)d_in[4];
  const float* drop1 = (const float*)d_in[5];
  const float* drop2 = (const float*)d_in[6];
  float* out = (float*)d_out;

  char* ws = (char*)d_ws;
  unsigned short* c1b  = (unsigned short*)(ws);                 // 8192*512 bf16 = 8 MiB
  unsigned int*   fnu  = (unsigned int*)(ws + 8388608);         // 8192*128 bf16 = 2 MiB
  float* M1    = (float*)(ws + 10485760);
  float* Z1    = (float*)(ws + 10485760 + 1 * 32768);
  float* invS1 = (float*)(ws + 10485760 + 2 * 32768);
  float* M2    = (float*)(ws + 10485760 + 3 * 32768);
  float* Z2    = (float*)(ws + 10485760 + 4 * 32768);
  float* invT  = (float*)(ws + 10485760 + 5 * 32768);
  float* diag  = (float*)(ws + 10485760 + 6 * 32768);

  gemm1_patchconv<<<dim3(64, 8), 256, 0, stream>>>(x, convw, convb, c1b);
  gemm2_fc<<<dim3(64, 2), 256, 0, stream>>>(c1b, fc1w, fc1b, out);
  rownorm<<<1024, 256, 0, stream>>>(out, fnu);
  gemm3_affinity<<<dim3(2, 4, 32), 256, 0, stream>>>(fnu, out + A_OUT_OFF);
  colstats<<<1024, 256, 0, stream>>>(out + A_OUT_OFF, drop1, M1, Z1, invS1);
  rowstats<<<1024, 256, 0, stream>>>(out + A_OUT_OFF, drop2, M2, Z2, invT);
  diagk<<<1024, 256, 0, stream>>>(out + A_OUT_OFF, drop1, drop2, M1, Z1, invS1, M2, Z2, invT, diag);
  lossk<<<1, 256, 0, stream>>>(diag, out);
}
